// MultiScaleDecoder_slow_11373073400241
// MI455X (gfx1250) — matrix operand order repaired and checked by reading, not compiled, not run on hardware
//
#include <hip/hip_runtime.h>
#include <hip/hip_bf16.h>
#include <math.h>

// Problem constants (from reference)
#define BB   256   // batch
#define ZD   128   // Z_DIM (fc output)
#define ED   256   // E_DIM
#define HH   512   // H_DIM
#define TT   2048  // T
#define NG   1536  // 3*H

typedef __attribute__((ext_vector_type(16))) __bf16 v16bf;
typedef __attribute__((ext_vector_type(8)))  __bf16 v8bf;
typedef __attribute__((ext_vector_type(8)))  float  v8f;

// ---------------------------------------------------------------------------
// WMMA helpers (CDNA5: D = A(16x32 bf16) x B(32x16 bf16) + C(16x16 f32))
// ---------------------------------------------------------------------------
__device__ __forceinline__ v8f wmma_bf16(v16bf a, v16bf b, v8f c) {
  // (neg_a, A, neg_b, B, c_mod, C, reuse_a, reuse_b)
  return __builtin_amdgcn_wmma_f32_16x16x32_bf16(false, a, false, b,
                                                 (short)0, c, false, false);
}

// A tile load: row-major bf16 matrix, row pointer `rowp` (this lane's M row).
// Lane l (half = l>>4): VGPR elems 0..7 = K kb+half*8 .. +7, elems 8..15 = +16..+23
__device__ __forceinline__ v16bf load_tileA(const __bf16* __restrict__ rowp,
                                            int kb, int half) {
  v16bf a;
  const __bf16* p = rowp + kb + half * 8;
  ((v8bf*)&a)[0] = *(const v8bf*)(p);
  ((v8bf*)&a)[1] = *(const v8bf*)(p + 16);
  return a;
}

// B tile load: B[k][n] = W[n_global][k] (row-major W, row pointer = this lane's N row).
// Lane l (half = l>>4): elems 0..7 = K kb+half*8 .. +7, elems 8..15 = +16..+23
__device__ __forceinline__ v16bf load_tileB(const __bf16* __restrict__ rowp,
                                            int kb, int half) {
  v16bf b;
  const __bf16* p = rowp + kb + half * 8;
  ((v8bf*)&b)[0] = *(const v8bf*)(p);
  ((v8bf*)&b)[1] = *(const v8bf*)(p + 16);
  return b;
}

__device__ __forceinline__ float sigmoidf_(float x) {
  return 1.0f / (1.0f + __expf(-x));
}

// ---------------------------------------------------------------------------
// One-time kernels
// ---------------------------------------------------------------------------

// gi = e @ w_ih.T + b_ih   (256 x 1536, K=256). One-time, simple.
__global__ void compute_gi_kernel(const float* __restrict__ e,
                                  const float* __restrict__ w_ih,
                                  const float* __restrict__ b_ih,
                                  float* __restrict__ gi) {
  int col = blockIdx.x;      // 0..1535
  int b   = threadIdx.x;     // 0..255
  const float* er = e + b * ED;
  const float* wr = w_ih + col * ED;
  float acc = b_ih[col];
  #pragma unroll 4
  for (int k = 0; k < ED; ++k) acc += er[k] * wr[k];
  gi[b * NG + col] = acc;
}

__global__ void f32_to_bf16_kernel(const float* __restrict__ src,
                                   __bf16* __restrict__ dst, int n) {
  int i = blockIdx.x * 256 + threadIdx.x;
  if (i < n) dst[i] = (__bf16)src[i];
}

__global__ void init_h_kernel(float* __restrict__ h32, __bf16* __restrict__ h16) {
  int i = blockIdx.x * 256 + threadIdx.x;   // n = 256*512 = 131072
  h32[i] = 0.0f;
  h16[i] = (__bf16)0.0f;
}

// ---------------------------------------------------------------------------
// Per-timestep kernel. Grid = 640 one-wave workgroups (32 threads):
//   blocks [0,512): GRU update  h[t] = GRU(h[t-1])        (skipped when t==TT)
//   blocks [512,640): FC        y[t-1] = h[t-1]@w_fc.T+b  (skipped when t==0)
// ---------------------------------------------------------------------------
__global__ void __launch_bounds__(32)
gru_step_kernel(const float*  __restrict__ gi,
                const float*  __restrict__ b_hh,
                const float*  __restrict__ h32_prev,
                float*        __restrict__ h32_cur,
                const __bf16* __restrict__ h16_prev,
                __bf16*       __restrict__ h16_cur,
                const __bf16* __restrict__ whh16,   // (1536, 512) bf16 row-major
                const __bf16* __restrict__ wfc16,   // (128, 512)  bf16 row-major
                const float*  __restrict__ b_fc,
                float*        __restrict__ out,     // (256, 128, 2048) f32
                int t) {
  const int blk  = blockIdx.x;
  const int l    = threadIdx.x & 31;
  const int half = l >> 4;     // 0 or 1
  const int lm   = l & 15;

  if (blk < 512) {
    // ---------------- GRU tile: 16 batch rows x 16 h-units ----------------
    if (t >= TT) return;                       // uniform; no GRU work on final launch
    const int m0 = (blk >> 5) * 16;            // batch tile
    const int j0 = (blk & 31) * 16;            // h-unit tile

    v8f accr = {}, accz = {}, accn = {};

    const __bf16* arow = h16_prev + (size_t)(m0 + lm) * HH;           // A: this lane's batch row
    const __bf16* wr_r = whh16 + (size_t)(j0 + lm) * HH;              // B rows per gate
    const __bf16* wr_z = whh16 + (size_t)(HH + j0 + lm) * HH;
    const __bf16* wr_n = whh16 + (size_t)(2 * HH + j0 + lm) * HH;

    #pragma unroll 4
    for (int kt = 0; kt < HH / 32; ++kt) {
      const int kb = kt * 32;
      v16bf a  = load_tileA(arow, kb, half);
      v16bf br = load_tileB(wr_r, kb, half);
      v16bf bz = load_tileB(wr_z, kb, half);
      v16bf bn = load_tileB(wr_n, kb, half);
      accr = wmma_bf16(a, br, accr);
      accz = wmma_bf16(a, bz, accz);
      accn = wmma_bf16(a, bn, accn);
    }

    // Epilogue: D elem v -> (M = m0 + v + 8*half, N = j0 + lm)
    const int jj   = j0 + lm;
    const float bhr = b_hh[jj];
    const float bhz = b_hh[HH + jj];
    const float bhn = b_hh[2 * HH + jj];
    #pragma unroll
    for (int v = 0; v < 8; ++v) {
      const int mm = m0 + v + half * 8;
      const size_t gbase = (size_t)mm * NG;
      const float pr = gi[gbase + jj]          + accr[v] + bhr;
      const float pz = gi[gbase + HH + jj]     + accz[v] + bhz;
      const float hn = accn[v] + bhn;
      const float gin = gi[gbase + 2 * HH + jj];
      const float r  = sigmoidf_(pr);
      const float zg = sigmoidf_(pz);
      const float nn = tanhf(gin + r * hn);
      const size_t hidx = (size_t)mm * HH + jj;
      const float hnew = (1.0f - zg) * nn + zg * h32_prev[hidx];
      h32_cur[hidx] = hnew;
      h16_cur[hidx] = (__bf16)hnew;
    }
  } else {
    // ---------------- FC tile: y[t-1] = h[t-1] @ w_fc.T + b_fc ------------
    if (t < 1) return;                         // uniform
    const int fid = blk - 512;                 // 0..127
    const int m0  = (fid >> 3) * 16;           // batch tile (16 tiles)
    const int n0  = (fid & 7) * 16;            // output-col tile (8 tiles)

    v8f acc = {};
    const __bf16* arow = h16_prev + (size_t)(m0 + lm) * HH;
    const __bf16* wrow = wfc16 + (size_t)(n0 + lm) * HH;

    #pragma unroll 4
    for (int kt = 0; kt < HH / 32; ++kt) {
      const int kb = kt * 32;
      v16bf a = load_tileA(arow, kb, half);
      v16bf b = load_tileB(wrow, kb, half);
      acc = wmma_bf16(a, b, acc);
    }

    const int nn = n0 + lm;
    const float bias = b_fc[nn];
    #pragma unroll
    for (int v = 0; v < 8; ++v) {
      const int bb = m0 + v + half * 8;
      out[((size_t)bb * ZD + nn) * TT + (t - 1)] = acc[v] + bias;
    }
  }
}

// ---------------------------------------------------------------------------
// Host-side orchestration (graph-capturable: stream launches only)
// ---------------------------------------------------------------------------
extern "C" void kernel_launch(void* const* d_in, const int* in_sizes, int n_in,
                              void* d_out, int out_size, void* d_ws, size_t ws_size,
                              hipStream_t stream) {
  // Input order: z_ctx, z, z_slow, z_fast, e, w_ih, w_hh, b_ih, b_hh, w_fc, b_fc
  const float* e    = (const float*)d_in[4];
  const float* w_ih = (const float*)d_in[5];
  const float* w_hh = (const float*)d_in[6];
  const float* b_ih = (const float*)d_in[7];
  const float* b_hh = (const float*)d_in[8];
  const float* w_fc = (const float*)d_in[9];
  const float* b_fc = (const float*)d_in[10];
  float* out = (float*)d_out;

  // Workspace layout (bytes, all 256B-aligned); total ~4.7 MB
  char* ws = (char*)d_ws;
  float*  gi    = (float*) (ws + 0);              // 256*1536*4 = 1,572,864
  float*  h32a  = (float*) (ws + 1572864);        // 256*512*4  =   524,288
  float*  h32b  = (float*) (ws + 2097152);        //              524,288
  __bf16* h16a  = (__bf16*)(ws + 2621440);        // 256*512*2  =   262,144
  __bf16* h16b  = (__bf16*)(ws + 2883584);        //              262,144
  __bf16* whh16 = (__bf16*)(ws + 3145728);        // 1536*512*2 = 1,572,864
  __bf16* wfc16 = (__bf16*)(ws + 4718592);        // 128*512*2  =   131,072

  // One-time prep: gi = e@w_ih.T + b_ih; bf16 weight copies; h0 = 0.
  compute_gi_kernel<<<NG, 256, 0, stream>>>(e, w_ih, b_ih, gi);
  f32_to_bf16_kernel<<<(NG * HH + 255) / 256, 256, 0, stream>>>(w_hh, whh16, NG * HH);
  f32_to_bf16_kernel<<<(ZD * HH + 255) / 256, 256, 0, stream>>>(w_fc, wfc16, ZD * HH);
  init_h_kernel<<<(BB * HH) / 256, 256, 0, stream>>>(h32a, h16a);

  // Sequential recurrence: step t computes h[t] (if t<T) and y[t-1] (if t>=1).
  for (int t = 0; t <= TT; ++t) {
    const bool even = (t & 1) == 0;
    const float*  hp32 = even ? h32a : h32b;
    float*        hc32 = even ? h32b : h32a;
    const __bf16* hp16 = even ? h16a : h16b;
    __bf16*       hc16 = even ? h16b : h16a;
    gru_step_kernel<<<640, 32, 0, stream>>>(gi, b_hh, hp32, hc32, hp16, hc16,
                                            whh16, wfc16, b_fc, out, t);
  }
}
